// InteractionPPBlock_16286515986996
// MI455X (gfx1250) — matrix operand order repaired and checked by reading, not compiled, not run on hardware
//
#include <hip/hip_runtime.h>

typedef __attribute__((ext_vector_type(16))) _Float16 v16h;
typedef __attribute__((ext_vector_type(8)))  float    v8f;

#define NSR 42
#define HSTR 132

__device__ __forceinline__ float silu_f(float v) {
  return v * (1.0f / (1.0f + __expf(-v)));
}

__device__ __forceinline__ v8f wmma_f16(v16h a, v16h b, v8f c) {
  return __builtin_amdgcn_wmma_f32_16x16x32_f16(false, a, false, b, (short)0, c,
                                                false, false);
}

// Pack fp32 weight W[K][N] into f16 B-fragment layout:
// frag (kb,nb): lane's v16h element e = W[kb*32 + (e>>3)*16 + (lane>>4)*8 + (e&7)][nb*16 + (lane&15)]
__device__ __forceinline__ void pack_b(_Float16* dst, const float* src, int K, int N, int tid) {
  const int NB = N >> 4;
  for (int d = tid; d < K * N; d += 256) {
    int e  = d & 15;
    int ln = (d >> 4) & 31;
    int fr = d >> 9;
    int nb = fr % NB, kb = fr / NB;
    int Ks = kb * 32 + (e >> 3) * 16 + (ln >> 4) * 8 + (e & 7);
    int Ns = nb * 16 + (ln & 15);
    dst[d] = (_Float16)src[Ks * N + Ns];
  }
}

__global__ void prep_kernel(
    const float* __restrict__ W_rbf1, const float* __restrict__ W_rbf2,
    const float* __restrict__ W_sbf1, const float* __restrict__ W_sbf2,
    const float* __restrict__ W_kj, const float* __restrict__ W_ji,
    const float* __restrict__ W_down, const float* __restrict__ W_up,
    const float* __restrict__ W_lin,
    const float* __restrict__ rbW1, const float* __restrict__ rbW2,
    const float* __restrict__ raW1, const float* __restrict__ raW2,
    _Float16* __restrict__ Wji_p, _Float16* __restrict__ Wkj_p,
    _Float16* __restrict__ Wdown_p, _Float16* __restrict__ Wup_p,
    _Float16* __restrict__ Wlin_p,
    _Float16* __restrict__ W1p, _Float16* __restrict__ W2p,
    _Float16* __restrict__ W3p, _Float16* __restrict__ W4p,
    _Float16* __restrict__ Wsbf_p, float* __restrict__ Wrbfc)
{
  const int tid = threadIdx.x;
  pack_b(Wji_p,   W_ji,   128, 128, tid);
  pack_b(Wkj_p,   W_kj,   128, 128, tid);
  pack_b(Wdown_p, W_down, 128,  64, tid);
  pack_b(Wup_p,   W_up,    64, 128, tid);
  pack_b(Wlin_p,  W_lin,  128, 128, tid);
  pack_b(W1p, rbW1, 128, 128, tid);
  pack_b(W2p, rbW2, 128, 128, tid);
  pack_b(W3p, raW1, 128, 128, tid);
  pack_b(W4p, raW2, 128, 128, tid);
  // Combined sbf projection (W_sbf1 @ W_sbf2) [42,64], zero-padded to K=64, packed.
  for (int d = tid; d < 64 * 64; d += 256) {
    int e  = d & 15;
    int ln = (d >> 4) & 31;
    int fr = d >> 9;
    int nb = fr & 3, kb = fr >> 2;
    int Ks = kb * 32 + (e >> 3) * 16 + (ln >> 4) * 8 + (e & 7);
    int Ns = nb * 16 + (ln & 15);
    float s = 0.f;
    if (Ks < NSR) {
      for (int j = 0; j < 8; ++j) s += W_sbf1[Ks * 8 + j] * W_sbf2[j * 64 + Ns];
    }
    Wsbf_p[d] = (_Float16)s;
  }
  // Combined rbf projection (W_rbf1 @ W_rbf2) [6,128] fp32 (used element-wise).
  for (int d = tid; d < 6 * 128; d += 256) {
    int r = d >> 7, n = d & 127;
    float s = 0.f;
    for (int j = 0; j < 8; ++j) s += W_rbf1[r * 8 + j] * W_rbf2[j * 128 + n];
    Wrbfc[d] = s;
  }
}

__global__ void zero_kernel(float4* __restrict__ p, long n4) {
  long i = (long)blockIdx.x * blockDim.x + threadIdx.x;
  if (i < n4) p[i] = make_float4(0.f, 0.f, 0.f, 0.f);
}

// Kernel A: per 16-edge tile (one wave): x_ji, x_kj*rbf_e, x_kj_down.
__global__ __launch_bounds__(32) void edge_fwd_kernel(
    const float* __restrict__ x, const float* __restrict__ rbf,
    const _Float16* __restrict__ Wji_p, const float* __restrict__ b_ji,
    const _Float16* __restrict__ Wkj_p, const float* __restrict__ b_kj,
    const _Float16* __restrict__ Wdown_p, const float* __restrict__ Wrbfc,
    _Float16* __restrict__ xji_out, _Float16* __restrict__ xkjd_out)
{
  const int lane = threadIdx.x;
  const int half = lane >> 4, lr = lane & 15;
  const int e0 = blockIdx.x * 16;

  __shared__ __align__(16) float    xs[16 * HSTR];
  __shared__ __align__(32) _Float16 ab[16 * 128];
  __shared__ float rbfs[16 * 6];

  for (int i = lane; i < 2048; i += 32) {
    int r = i >> 7, c = i & 127;
    xs[r * HSTR + c] = x[(size_t)e0 * 128 + i];
  }
  for (int i = lane; i < 96; i += 32) rbfs[i] = rbf[(size_t)e0 * 6 + i];
  __syncthreads();

  v16h ax[4];
  #pragma unroll
  for (int kb = 0; kb < 4; ++kb) {
    v16h a;
    #pragma unroll
    for (int e = 0; e < 16; ++e) {
      int K = kb * 32 + (e >> 3) * 16 + half * 8 + (e & 7);
      a[e] = (_Float16)xs[lr * HSTR + K];
    }
    ax[kb] = a;
  }

  #pragma unroll
  for (int nb = 0; nb < 8; ++nb) {
    v8f c = {};
    #pragma unroll
    for (int kb = 0; kb < 4; ++kb)
      c = wmma_f16(ax[kb], ((const v16h*)Wji_p)[(kb * 8 + nb) * 32 + lane], c);
    float bs = b_ji[nb * 16 + lr];
    #pragma unroll
    for (int v = 0; v < 8; ++v) {
      int M = v + half * 8;
      xji_out[(size_t)(e0 + M) * 128 + nb * 16 + lr] = (_Float16)silu_f(c[v] + bs);
    }
  }

  #pragma unroll
  for (int nb = 0; nb < 8; ++nb) {
    v8f c = {};
    #pragma unroll
    for (int kb = 0; kb < 4; ++kb)
      c = wmma_f16(ax[kb], ((const v16h*)Wkj_p)[(kb * 8 + nb) * 32 + lane], c);
    const int N = nb * 16 + lr;
    float bs = b_kj[N];
    float wr[6];
    #pragma unroll
    for (int r = 0; r < 6; ++r) wr[r] = Wrbfc[r * 128 + N];
    #pragma unroll
    for (int v = 0; v < 8; ++v) {
      int M = v + half * 8;
      float re = 0.f;
      #pragma unroll
      for (int r = 0; r < 6; ++r) re += rbfs[M * 6 + r] * wr[r];
      float y = silu_f(c[v] + bs) * re;
      int kk = N & 31;
      ab[(((N >> 5) * 32) + ((kk >> 3) & 1) * 16 + M) * 16 +
         ((kk >> 4) * 8 + (kk & 7))] = (_Float16)y;
    }
  }
  __syncthreads();

  v16h ad[4];
  #pragma unroll
  for (int kb = 0; kb < 4; ++kb) ad[kb] = ((const v16h*)ab)[kb * 32 + lane];
  #pragma unroll
  for (int nb = 0; nb < 4; ++nb) {
    v8f c = {};
    #pragma unroll
    for (int kb = 0; kb < 4; ++kb)
      c = wmma_f16(ad[kb], ((const v16h*)Wdown_p)[(kb * 4 + nb) * 32 + lane], c);
    #pragma unroll
    for (int v = 0; v < 8; ++v) {
      int M = v + half * 8;
      xkjd_out[(size_t)(e0 + M) * 64 + nb * 16 + lr] = (_Float16)silu_f(c[v]);
    }
  }
}

// Kernel B: per 16-triplet tile: sbf_e via WMMA, gather x_kj_down, scatter-add agg.
__global__ __launch_bounds__(32) void triplet_kernel(
    const float* __restrict__ sbf,
    const int* __restrict__ idx_kj, const int* __restrict__ idx_ji,
    const _Float16* __restrict__ Wsbf_p, const _Float16* __restrict__ xkjd,
    float* __restrict__ agg)
{
  const int lane = threadIdx.x;
  const int half = lane >> 4, lr = lane & 15;
  const int t0 = blockIdx.x * 16;

  __shared__ float ss[16 * NSR];
  __shared__ int ikj[16], iji[16];

  for (int i = lane; i < 16 * NSR; i += 32) ss[i] = sbf[(size_t)t0 * NSR + i];
  if (lane < 16) {
    ikj[lane] = idx_kj[t0 + lane] * 64;
    iji[lane] = idx_ji[t0 + lane] * 64;
  }
  __syncthreads();

  v16h a[2];
  #pragma unroll
  for (int kb = 0; kb < 2; ++kb) {
    v16h t;
    #pragma unroll
    for (int e = 0; e < 16; ++e) {
      int K = kb * 32 + (e >> 3) * 16 + half * 8 + (e & 7);
      t[e] = (K < NSR) ? (_Float16)ss[lr * NSR + K] : (_Float16)0.f;
    }
    a[kb] = t;
  }
  int mk[8], mj[8];
  #pragma unroll
  for (int v = 0; v < 8; ++v) {
    int M = v + half * 8;
    mk[v] = ikj[M];
    mj[v] = iji[M];
  }

  #pragma unroll
  for (int nb = 0; nb < 4; ++nb) {
    v8f c = {};
    #pragma unroll
    for (int kb = 0; kb < 2; ++kb)
      c = wmma_f16(a[kb], ((const v16h*)Wsbf_p)[(kb * 4 + nb) * 32 + lane], c);
    const int ch = nb * 16 + lr;
    #pragma unroll
    for (int v = 0; v < 8; ++v) {
      float g = (float)xkjd[(size_t)(mk[v] + ch)];
      atomicAdd(&agg[(size_t)(mj[v] + ch)], g * c[v]);
    }
  }
}

enum { MODE_RES1 = 0, MODE_RES2 = 1, MODE_LIN = 2 };

template <int MODE>
__device__ __forceinline__ void gemm_stage128(
    const _Float16* __restrict__ Wp, const float* __restrict__ bias,
    _Float16* ab, float* hb, const float* __restrict__ xrow, int lane)
{
  const int half = lane >> 4, lr = lane & 15;
  __syncthreads();
  v16h a[4];
  #pragma unroll
  for (int kb = 0; kb < 4; ++kb) a[kb] = ((const v16h*)ab)[kb * 32 + lane];
  __syncthreads();
  #pragma unroll
  for (int nb = 0; nb < 8; ++nb) {
    v8f c = {};
    #pragma unroll
    for (int kb = 0; kb < 4; ++kb)
      c = wmma_f16(a[kb], ((const v16h*)Wp)[(kb * 8 + nb) * 32 + lane], c);
    float bs = bias[nb * 16 + lr];
    #pragma unroll
    for (int v = 0; v < 8; ++v) {
      int M = v + half * 8, N = nb * 16 + lr;
      float y = silu_f(c[v] + bs);
      if (MODE == MODE_RES2) y += hb[M * HSTR + N];
      if (MODE == MODE_LIN)  y += xrow[(size_t)M * 128 + N];
      if (MODE != MODE_RES1) hb[M * HSTR + N] = y;
      int kk = N & 31;
      ab[(((N >> 5) * 32) + ((kk >> 3) & 1) * 16 + M) * 16 +
         ((kk >> 4) * 8 + (kk & 7))] = (_Float16)y;
    }
  }
}

__global__ __launch_bounds__(32) void edge_out_kernel(
    const float* __restrict__ x, const float* __restrict__ agg,
    const _Float16* __restrict__ xji,
    const _Float16* __restrict__ Wup_p,
    const _Float16* __restrict__ W1p, const float* __restrict__ b1,
    const _Float16* __restrict__ W2p, const float* __restrict__ b2,
    const _Float16* __restrict__ Wlinp, const float* __restrict__ blin,
    const _Float16* __restrict__ W3p, const float* __restrict__ b3,
    const _Float16* __restrict__ W4p, const float* __restrict__ b4,
    float* __restrict__ out)
{
  const int lane = threadIdx.x;
  const int half = lane >> 4, lr = lane & 15;
  const int e0 = blockIdx.x * 16;

  __shared__ __align__(16) float    hb[16 * HSTR];
  __shared__ __align__(32) _Float16 ab[16 * 128];

  for (int i = lane; i < 1024; i += 32) {
    int r = i >> 6, cc = i & 63;
    hb[r * 68 + cc] = agg[(size_t)e0 * 64 + i];
  }
  __syncthreads();
  v16h a2[2];
  #pragma unroll
  for (int kb = 0; kb < 2; ++kb) {
    v16h t;
    #pragma unroll
    for (int e = 0; e < 16; ++e) {
      int K = kb * 32 + (e >> 3) * 16 + half * 8 + (e & 7);
      t[e] = (_Float16)hb[lr * 68 + K];
    }
    a2[kb] = t;
  }
  __syncthreads();

  #pragma unroll
  for (int nb = 0; nb < 8; ++nb) {
    v8f c = {};
    #pragma unroll
    for (int kb = 0; kb < 2; ++kb)
      c = wmma_f16(a2[kb], ((const v16h*)Wup_p)[(kb * 8 + nb) * 32 + lane], c);
    #pragma unroll
    for (int v = 0; v < 8; ++v) {
      int M = v + half * 8, N = nb * 16 + lr;
      float y = silu_f(c[v]) + (float)xji[(size_t)(e0 + M) * 128 + N];
      hb[M * HSTR + N] = y;
      int kk = N & 31;
      ab[(((N >> 5) * 32) + ((kk >> 3) & 1) * 16 + M) * 16 +
         ((kk >> 4) * 8 + (kk & 7))] = (_Float16)y;
    }
  }

  const float* xrow = x + (size_t)e0 * 128;
  gemm_stage128<MODE_RES1>(W1p,   b1,   ab, hb, xrow, lane);
  gemm_stage128<MODE_RES2>(W2p,   b2,   ab, hb, xrow, lane);
  gemm_stage128<MODE_LIN >(Wlinp, blin, ab, hb, xrow, lane);
  gemm_stage128<MODE_RES1>(W3p,   b3,   ab, hb, xrow, lane);
  gemm_stage128<MODE_RES2>(W4p,   b4,   ab, hb, xrow, lane);
  __syncthreads();

  for (int i = lane; i < 2048; i += 32) {
    int r = i >> 7, cc = i & 127;
    out[(size_t)e0 * 128 + i] = hb[r * HSTR + cc];
  }
}

extern "C" void kernel_launch(void* const* d_in, const int* in_sizes, int n_in,
                              void* d_out, int out_size, void* d_ws, size_t ws_size,
                              hipStream_t stream) {
  const float* x      = (const float*)d_in[0];
  const float* rbf    = (const float*)d_in[1];
  const float* sbf    = (const float*)d_in[2];
  const float* W_rbf1 = (const float*)d_in[3];
  const float* W_rbf2 = (const float*)d_in[4];
  const float* W_sbf1 = (const float*)d_in[5];
  const float* W_sbf2 = (const float*)d_in[6];
  const float* W_kj   = (const float*)d_in[7];
  const float* b_kj   = (const float*)d_in[8];
  const float* W_ji   = (const float*)d_in[9];
  const float* b_ji   = (const float*)d_in[10];
  const float* W_down = (const float*)d_in[11];
  const float* W_up   = (const float*)d_in[12];
  const float* W_lin  = (const float*)d_in[13];
  const float* b_lin  = (const float*)d_in[14];
  const float* rb_W1  = (const float*)d_in[15];
  const float* rb_b1  = (const float*)d_in[16];
  const float* rb_W2  = (const float*)d_in[17];
  const float* rb_b2  = (const float*)d_in[18];
  const float* ra_W1  = (const float*)d_in[19];
  const float* ra_b1  = (const float*)d_in[20];
  const float* ra_W2  = (const float*)d_in[21];
  const float* ra_b2  = (const float*)d_in[22];
  const int* idx_kj   = (const int*)d_in[23];
  const int* idx_ji   = (const int*)d_in[24];
  float* out = (float*)d_out;

  const int E = in_sizes[0] / 128;
  const int T = in_sizes[2] / NSR;

  char* ws = (char*)d_ws;
  size_t off = 0;
  auto alloc = [&](size_t bytes) -> char* {
    char* p = ws + off;
    off = (off + bytes + 255) & ~(size_t)255;
    return p;
  };
  _Float16* Wji_p   = (_Float16*)alloc(128 * 128 * 2);
  _Float16* Wkj_p   = (_Float16*)alloc(128 * 128 * 2);
  _Float16* Wdown_p = (_Float16*)alloc(128 * 64 * 2);
  _Float16* Wup_p   = (_Float16*)alloc(64 * 128 * 2);
  _Float16* Wlin_p  = (_Float16*)alloc(128 * 128 * 2);
  _Float16* W1p     = (_Float16*)alloc(128 * 128 * 2);
  _Float16* W2p     = (_Float16*)alloc(128 * 128 * 2);
  _Float16* W3p     = (_Float16*)alloc(128 * 128 * 2);
  _Float16* W4p     = (_Float16*)alloc(128 * 128 * 2);
  _Float16* Wsbf_p  = (_Float16*)alloc(64 * 64 * 2);
  float*    Wrbfc   = (float*)   alloc(6 * 128 * 4);
  _Float16* xji     = (_Float16*)alloc((size_t)E * 128 * 2);
  _Float16* xkjd    = (_Float16*)alloc((size_t)E * 64 * 2);
  float*    agg     = (float*)   alloc((size_t)E * 64 * 4);

  prep_kernel<<<1, 256, 0, stream>>>(
      W_rbf1, W_rbf2, W_sbf1, W_sbf2, W_kj, W_ji, W_down, W_up, W_lin,
      rb_W1, rb_W2, ra_W1, ra_W2,
      Wji_p, Wkj_p, Wdown_p, Wup_p, Wlin_p, W1p, W2p, W3p, W4p, Wsbf_p, Wrbfc);

  long n4 = (long)E * 64 / 4;
  zero_kernel<<<(int)((n4 + 255) / 256), 256, 0, stream>>>((float4*)agg, n4);

  edge_fwd_kernel<<<E / 16, 32, 0, stream>>>(
      x, rbf, Wji_p, b_ji, Wkj_p, b_kj, Wdown_p, Wrbfc, xji, xkjd);

  triplet_kernel<<<T / 16, 32, 0, stream>>>(sbf, idx_kj, idx_ji, Wsbf_p, xkjd, agg);

  edge_out_kernel<<<E / 16, 32, 0, stream>>>(
      x, agg, xji, Wup_p, W1p, rb_b1, W2p, rb_b2, Wlin_p, b_lin,
      W3p, ra_b1, W4p, ra_b2, out);

  (void)n_in; (void)out_size; (void)ws_size;
}
